// ContActor_88776974008444
// MI455X (gfx1250) — matrix operand order repaired and checked by reading, not compiled, not run on hardware
//
#include <hip/hip_runtime.h>
#include <hip/hip_bf16.h>
#include <math.h>

// ---------------------------------------------------------------------------
// CDNA5 (gfx1250, wave32) WMMA: D(f32 16x16) = A(bf16 16x32) x B(bf16 32x16) + C
// ---------------------------------------------------------------------------
typedef __bf16 bf16_t;
typedef bf16_t v16bf __attribute__((ext_vector_type(16)));
typedef bf16_t v8bf  __attribute__((ext_vector_type(8)));
typedef float  v8f   __attribute__((ext_vector_type(8)));

#define B_   64
#define N_   1024
#define DIN_ 16
#define D_   256
#define H_   4
#define L_   3
#define DH_  64
#define DFF_ 1024
#define K_   64

__device__ inline v8f zero_v8f() {
  v8f r;
#pragma unroll
  for (int i = 0; i < 8; ++i) r[i] = 0.0f;
  return r;
}

__device__ inline v16bf mk16(v8bf lo, v8bf hi) {
  v16bf r;
#pragma unroll
  for (int i = 0; i < 8; ++i) { r[i] = lo[i]; r[i + 8] = hi[i]; }
  return r;
}

// A-matrix fragment (16x32 bf16, M x K), ISA 7.12.2 layout:
// lanes 0-15: row=lane,    K = {0..7} then {16..23}
// lanes16-31: row=lane-16, K = {8..15} then {24..31}
__device__ inline v16bf frag_a(const bf16_t* tile, int stride, int row_base, int kk, int lane) {
  const bf16_t* p = tile + (size_t)(row_base + (lane & 15)) * stride + kk + ((lane & 16) ? 8 : 0);
  return mk16(*(const v8bf*)p, *(const v8bf*)(p + 16));
}

// B-matrix fragment (32x16 bf16, K x N) from K-contiguous transposed tile bt[col][k]:
// lanes 0-15: col=lane, K={0..7} then {16..23}; lanes16-31: col=lane-16, K={8..15} then {24..31}.
__device__ inline v16bf frag_b(const bf16_t* tile, int stride, int col_base, int kk, int lane) {
  const bf16_t* p = tile + (size_t)(col_base + (lane & 15)) * stride + kk + ((lane & 16) ? 8 : 0);
  return mk16(*(const v8bf*)p, *(const v8bf*)(p + 16));
}

#define WMMA_BF16(A, Bf, C) \
  __builtin_amdgcn_wmma_f32_16x16x32_bf16(false, (A), false, (Bf), (short)0, (C), false, false)

// Async global->LDS copy of 16 bytes per lane (GLOBAL_LOAD_ASYNC_TO_LDS_B128,
// ASYNCcnt-tracked). lds_off is the wave-relative LDS byte address (low 32 bits
// of a generic pointer to __shared__), gaddr a 16B-aligned global address.
__device__ inline void async_ld_b128(unsigned lds_off, const bf16_t* gaddr) {
  asm volatile("global_load_async_to_lds_b128 %0, %1, off"
               :: "v"(lds_off), "v"(gaddr) : "memory");
}
__device__ inline void wait_async0() {
  asm volatile("s_wait_asynccnt 0x0" ::: "memory");
}

// ---------------------------------------------------------------------------
// fp32 -> bf16 bulk convert (weights, once per launch)
// ---------------------------------------------------------------------------
__global__ __launch_bounds__(256) void cvt_kernel(const float* __restrict__ src,
                                                  bf16_t* __restrict__ dst, int n) {
  int i = blockIdx.x * 256 + threadIdx.x;
  if (i < n) dst[i] = (bf16_t)src[i];
}

// ---------------------------------------------------------------------------
// GEMM: out = op(A_bf16[M,K] @ W_bf16[K,Nc] + bias) (+ resid_f32).
// Outputs optionally fp32 (Cf) and/or bf16 scaled (Cb, val*bfscale).
// Block tile 128x128, BK=64, 8 waves of 32x64. A-tile staged with async
// global->LDS B128 copies (no VGPR round-trip); W staged transposed.
// ---------------------------------------------------------------------------
#define GBM 128
#define GBN 128
#define GBK 64

__global__ __launch_bounds__(256) void gemm_wmma_kernel(
    const bf16_t* __restrict__ A, const bf16_t* __restrict__ W,
    const float* __restrict__ bias, const float* __restrict__ resid,
    float* __restrict__ Cf, bf16_t* __restrict__ Cb,
    int M, int K, int Nc, int relu, float bfscale) {
  __shared__ __align__(16) bf16_t a_lds[GBM][GBK];
  __shared__ __align__(16) bf16_t bt_lds[GBN][GBK];

  const int t = threadIdx.x;
  const int lane = t & 31, wv = t >> 5;
  const int wm = wv & 3, wn = wv >> 2;           // 4 x 2 wave grid -> 32x64 per wave
  const int m0 = blockIdx.y * GBM, n0 = blockIdx.x * GBN;
  const unsigned a_off = (unsigned)(size_t)&a_lds[0][0];

  v8f acc[2][4];
#pragma unroll
  for (int mt = 0; mt < 2; ++mt)
#pragma unroll
    for (int nt = 0; nt < 4; ++nt) acc[mt][nt] = zero_v8f();

  for (int k0 = 0; k0 < K; k0 += GBK) {
    // A tile 128x64 bf16: 8 x 16B chunks per row, hardware async copy.
#pragma unroll
    for (int i = 0; i < 4; ++i) {
      int idx = t + i * 256;                     // 0..1023
      int r = idx >> 3, c = idx & 7;
      if (m0 + r < M)
        async_ld_b128(a_off + (unsigned)((r * GBK + c * 8) * 2),
                      A + (size_t)(m0 + r) * K + k0 + c * 8);
    }
    // W tile transposed: bt[n][k] <- W[k0+k][n0+n], bf16 source.
#pragma unroll
    for (int i = 0; i < 16; ++i) {
      int idx = t + i * 256;                     // 0..4095
      int n = (idx & 63) * 2, kq = idx >> 6;
      const bf16_t* wp = W + (size_t)(k0 + kq) * Nc + n0 + n;
      bt_lds[n][kq] = wp[0];
      bt_lds[n + 1][kq] = wp[1];
    }
    wait_async0();
    __syncthreads();

#pragma unroll
    for (int kk = 0; kk < GBK; kk += 32) {
      v16bf af[2];
#pragma unroll
      for (int mt = 0; mt < 2; ++mt)
        af[mt] = frag_a(&a_lds[0][0], GBK, wm * 32 + mt * 16, kk, lane);
#pragma unroll
      for (int nt = 0; nt < 4; ++nt) {
        v16bf bfr = frag_b(&bt_lds[0][0], GBK, wn * 64 + nt * 16, kk, lane);
#pragma unroll
        for (int mt = 0; mt < 2; ++mt)
          acc[mt][nt] = WMMA_BF16(af[mt], bfr, acc[mt][nt]);
      }
    }
    __syncthreads();
  }

#pragma unroll
  for (int mt = 0; mt < 2; ++mt)
#pragma unroll
    for (int nt = 0; nt < 4; ++nt)
#pragma unroll
      for (int i = 0; i < 8; ++i) {
        int row = m0 + wm * 32 + mt * 16 + i + ((lane & 16) ? 8 : 0);
        int col = n0 + wn * 64 + nt * 16 + (lane & 15);
        if (row < M) {
          float v = acc[mt][nt][i];
          if (bias)  v += bias[col];
          if (relu)  v = fmaxf(v, 0.0f);
          if (resid) v += resid[(size_t)row * Nc + col];
          if (Cf)    Cf[(size_t)row * Nc + col] = v;
          if (Cb)    Cb[(size_t)row * Nc + col] = (bf16_t)(v * bfscale);
        }
      }
}

// ---------------------------------------------------------------------------
// Flash attention on bf16 q/k/v (q pre-scaled by 1/sqrt(DH) in its GEMM
// epilogue). One block per (b, head, 128-query tile). Q/K tiles staged with
// async global->LDS copies; V transposed manually. bf16 output.
// ---------------------------------------------------------------------------
__global__ __launch_bounds__(256) void attn_kernel(
    const bf16_t* __restrict__ Q, const bf16_t* __restrict__ Kb,
    const bf16_t* __restrict__ Vb, bf16_t* __restrict__ O) {
  __shared__ __align__(16) bf16_t q_lds[128][64];
  __shared__ __align__(16) bf16_t kt_lds[64][64];    // [key][feat] -> Bt for Q.K^T
  __shared__ __align__(16) bf16_t vt_lds[64][64];    // [feat][key] -> Bt for P.V
  __shared__ __align__(16) bf16_t p_lds[8][16][64];  // per-wave P tile

  const int t = threadIdx.x, lane = t & 31, wv = t >> 5;
  const int bid = blockIdx.x;
  const int qb = bid & 7, hh = (bid >> 3) & 3, b = bid >> 5;
  const size_t base = (size_t)b * N_ * D_ + (size_t)hh * DH_;
  const unsigned q_off = (unsigned)(size_t)&q_lds[0][0];
  const unsigned k_off = (unsigned)(size_t)&kt_lds[0][0];

  // Q tile (128x64 bf16, row stride D_): pure async copy.
#pragma unroll
  for (int i = 0; i < 4; ++i) {
    int idx = t + i * 256;
    int r = idx >> 3, c = idx & 7;
    async_ld_b128(q_off + (unsigned)((r * 64 + c * 8) * 2),
                  Q + base + (size_t)(qb * 128 + r) * D_ + c * 8);
  }

  float m_run[8], l_run[8];
  v8f acc_o[4];
#pragma unroll
  for (int i = 0; i < 8; ++i) { m_run[i] = -3.0e38f; l_run[i] = 0.0f; }
#pragma unroll
  for (int ft = 0; ft < 4; ++ft) acc_o[ft] = zero_v8f();

  for (int j = 0; j < N_ / 64; ++j) {
    __syncthreads();
    // K tile (64x64, direct image): async copy.
#pragma unroll
    for (int i = 0; i < 2; ++i) {
      int idx = t + i * 256;
      int r = idx >> 3, c = idx & 7;
      async_ld_b128(k_off + (unsigned)((r * 64 + c * 8) * 2),
                    Kb + base + (size_t)(j * 64 + r) * D_ + c * 8);
    }
    // V tile transposed: vt[f][key] <- V[key][f].
#pragma unroll
    for (int i = 0; i < 8; ++i) {
      int idx = t + i * 256;                    // 0..2047
      int key = idx >> 5, f = (idx & 31) * 2;
      const bf16_t* vp = Vb + base + (size_t)(j * 64 + key) * D_ + f;
      vt_lds[f][key] = vp[0];
      vt_lds[f + 1][key] = vp[1];
    }
    wait_async0();
    __syncthreads();

    // S = Qs @ K^T : 16 rows x 64 keys per wave.
    v8f s[4];
#pragma unroll
    for (int nt = 0; nt < 4; ++nt) s[nt] = zero_v8f();
#pragma unroll
    for (int kk = 0; kk < 64; kk += 32) {
      v16bf aq = frag_a(&q_lds[0][0], 64, wv * 16, kk, lane);
#pragma unroll
      for (int nt = 0; nt < 4; ++nt) {
        v16bf bk = frag_b(&kt_lds[0][0], 64, nt * 16, kk, lane);
        s[nt] = WMMA_BF16(aq, bk, s[nt]);
      }
    }

    // Online softmax. C layout: row = i + 8*(lane>=16), col = lane&15.
    const int prow = ((lane & 16) ? 8 : 0);
    const int pcol = lane & 15;
#pragma unroll
    for (int i = 0; i < 8; ++i) {
      float mx = fmaxf(fmaxf(s[0][i], s[1][i]), fmaxf(s[2][i], s[3][i]));
      mx = fmaxf(mx, __shfl_xor(mx, 8, 16));
      mx = fmaxf(mx, __shfl_xor(mx, 4, 16));
      mx = fmaxf(mx, __shfl_xor(mx, 2, 16));
      mx = fmaxf(mx, __shfl_xor(mx, 1, 16));
      float m_new = fmaxf(m_run[i], mx);
      float alpha = __expf(m_run[i] - m_new);
      float ps = 0.0f;
#pragma unroll
      for (int nt = 0; nt < 4; ++nt) {
        float p = __expf(s[nt][i] - m_new);
        ps += p;
        p_lds[wv][i + prow][nt * 16 + pcol] = (bf16_t)p;
      }
      ps += __shfl_xor(ps, 8, 16);
      ps += __shfl_xor(ps, 4, 16);
      ps += __shfl_xor(ps, 2, 16);
      ps += __shfl_xor(ps, 1, 16);
      l_run[i] = l_run[i] * alpha + ps;
      m_run[i] = m_new;
#pragma unroll
      for (int ft = 0; ft < 4; ++ft) acc_o[ft][i] *= alpha;
    }

    // Same-wave LDS RAW (P writes -> A-fragment reads).
    asm volatile("s_wait_dscnt 0x0" ::: "memory");

    // O += P @ V
#pragma unroll
    for (int kk = 0; kk < 64; kk += 32) {
      v16bf ap = frag_a(&p_lds[wv][0][0], 64, 0, kk, lane);
#pragma unroll
      for (int ft = 0; ft < 4; ++ft) {
        v16bf bv = frag_b(&vt_lds[0][0], 64, ft * 16, kk, lane);
        acc_o[ft] = WMMA_BF16(ap, bv, acc_o[ft]);
      }
    }
  }

#pragma unroll
  for (int ft = 0; ft < 4; ++ft)
#pragma unroll
    for (int i = 0; i < 8; ++i) {
      int row = qb * 128 + wv * 16 + i + ((lane & 16) ? 8 : 0);
      int col = ft * 16 + (lane & 15);
      O[base + (size_t)row * D_ + col] = (bf16_t)(acc_o[ft][i] / l_run[i]);
    }
}

// ---------------------------------------------------------------------------
// Embedding: h = x @ emb_W + b (K=16, VALU); emits fp32 master + bf16 copy.
// ---------------------------------------------------------------------------
__global__ __launch_bounds__(256) void embed_kernel(
    const float* __restrict__ x, const float* __restrict__ W,
    const float* __restrict__ bias, float* __restrict__ h,
    bf16_t* __restrict__ hb) {
  const int row = blockIdx.x, t = threadIdx.x;
  __shared__ float xr[DIN_];
  if (t < DIN_) xr[t] = x[(size_t)row * DIN_ + t];
  __syncthreads();
  float acc = bias[t];
#pragma unroll
  for (int k = 0; k < DIN_; ++k) acc += xr[k] * W[k * D_ + t];
  h[(size_t)row * D_ + t] = acc;
  hb[(size_t)row * D_ + t] = (bf16_t)acc;
}

// ---------------------------------------------------------------------------
// LayerNorm over D=256; emits fp32 master + bf16 copy.
// ---------------------------------------------------------------------------
__global__ __launch_bounds__(256) void ln_kernel(
    const float* __restrict__ x, const float* __restrict__ g,
    const float* __restrict__ bb, float* __restrict__ out,
    bf16_t* __restrict__ outb) {
  const int row = blockIdx.x, t = threadIdx.x;
  __shared__ float red[256];
  float v = x[(size_t)row * D_ + t];
  red[t] = v;
  __syncthreads();
  for (int s = 128; s > 0; s >>= 1) { if (t < s) red[t] += red[t + s]; __syncthreads(); }
  float mean = red[0] * (1.0f / D_);
  __syncthreads();
  float d = v - mean;
  red[t] = d * d;
  __syncthreads();
  for (int s = 128; s > 0; s >>= 1) { if (t < s) red[t] += red[t + s]; __syncthreads(); }
  float var = red[0] * (1.0f / D_);
  float r = d * rsqrtf(var + 1e-5f) * g[t] + bb[t];
  out[(size_t)row * D_ + t] = r;
  outb[(size_t)row * D_ + t] = (bf16_t)r;
}

// ---------------------------------------------------------------------------
// Mean over N -> bf16 h_mean (only consumed as GEMM A operand).
// ---------------------------------------------------------------------------
__global__ __launch_bounds__(256) void mean_kernel(const float* __restrict__ h,
                                                   bf16_t* __restrict__ hmb) {
  const int b = blockIdx.x, t = threadIdx.x;
  float acc = 0.0f;
  for (int n = 0; n < N_; ++n) acc += h[((size_t)b * N_ + n) * D_ + t];
  hmb[b * D_ + t] = (bf16_t)(acc * (1.0f / N_));
}

// ---------------------------------------------------------------------------
// weights = softmax((qv.kv/16 + gumbel)/tau) per batch row of N=1024.
// ---------------------------------------------------------------------------
__global__ __launch_bounds__(256) void score_softmax_kernel(
    const float* __restrict__ qv, const float* __restrict__ kv,
    const float* __restrict__ gumbel, float* __restrict__ w_out) {
  const int b = blockIdx.x, t = threadIdx.x;
  __shared__ float qs[D_];
  __shared__ float sv[N_];
  __shared__ float red[256];
  qs[t] = qv[b * D_ + t];
  __syncthreads();
  float lmax = -3.0e38f;
#pragma unroll
  for (int r = 0; r < 4; ++r) {
    int n = t + r * 256;
    const float* kr = kv + ((size_t)b * N_ + n) * D_;
    float acc = 0.0f;
    for (int k = 0; k < D_; ++k) acc += qs[k] * kr[k];
    float val = (acc * (1.0f / 16.0f) + gumbel[b * N_ + n]) * 2.0f;  // /tau, tau=0.5
    sv[n] = val;
    lmax = fmaxf(lmax, val);
  }
  red[t] = lmax;
  __syncthreads();
  for (int s = 128; s > 0; s >>= 1) { if (t < s) red[t] = fmaxf(red[t], red[t + s]); __syncthreads(); }
  float mx = red[0];
  __syncthreads();
  float lsum = 0.0f;
#pragma unroll
  for (int r = 0; r < 4; ++r) {
    int n = t + r * 256;
    float e = __expf(sv[n] - mx);
    sv[n] = e;
    lsum += e;
  }
  red[t] = lsum;
  __syncthreads();
  for (int s = 128; s > 0; s >>= 1) { if (t < s) red[t] += red[t + s]; __syncthreads(); }
  float inv = 1.0f / red[0];
#pragma unroll
  for (int r = 0; r < 4; ++r) {
    int n = t + r * 256;
    w_out[(size_t)b * N_ + n] = sv[n] * inv;
  }
}

// ---------------------------------------------------------------------------
// Iterative top-K (K=64) argmax, lowest-index tie-break (lax.top_k order).
// ---------------------------------------------------------------------------
__global__ __launch_bounds__(256) void topk_kernel(const float* __restrict__ w,
                                                   float* __restrict__ idx_f,
                                                   int* __restrict__ idx_i) {
  const int b = blockIdx.x, t = threadIdx.x;
  __shared__ float vals[N_];
  __shared__ float rv[256];
  __shared__ int ri[256];
#pragma unroll
  for (int r = 0; r < 4; ++r) vals[t + r * 256] = w[(size_t)b * N_ + t + r * 256];
  __syncthreads();
  for (int it = 0; it < K_; ++it) {
    float bv = -3.0e38f;
    int bi = 0x7fffffff;
#pragma unroll
    for (int r = 0; r < 4; ++r) {
      int n = t + r * 256;
      float v = vals[n];
      if (v > bv) { bv = v; bi = n; }
    }
    rv[t] = bv;
    ri[t] = bi;
    __syncthreads();
    for (int s = 128; s > 0; s >>= 1) {
      if (t < s) {
        if (rv[t + s] > rv[t] || (rv[t + s] == rv[t] && ri[t + s] < ri[t])) {
          rv[t] = rv[t + s];
          ri[t] = ri[t + s];
        }
      }
      __syncthreads();
    }
    if (t == 0) {
      int bidx = ri[0];
      idx_f[b * K_ + it] = (float)bidx;
      idx_i[b * K_ + it] = bidx;
      vals[bidx] = -3.0e38f;
    }
    __syncthreads();
  }
}

// ---------------------------------------------------------------------------
// alloc = softmax(logits); sel = renormalized gather at topk indices.
// ---------------------------------------------------------------------------
__global__ __launch_bounds__(256) void sel_kernel(const float* __restrict__ logits,
                                                  const int* __restrict__ tki,
                                                  float* __restrict__ selo) {
  const int b = blockIdx.x, t = threadIdx.x;
  __shared__ float sv[N_];
  __shared__ float red[256];
  __shared__ float gath[K_];
  float lmax = -3.0e38f;
#pragma unroll
  for (int r = 0; r < 4; ++r) {
    float v = logits[(size_t)b * N_ + t + r * 256];
    sv[t + r * 256] = v;
    lmax = fmaxf(lmax, v);
  }
  red[t] = lmax;
  __syncthreads();
  for (int s = 128; s > 0; s >>= 1) { if (t < s) red[t] = fmaxf(red[t], red[t + s]); __syncthreads(); }
  float mx = red[0];
  __syncthreads();
  float lsum = 0.0f;
#pragma unroll
  for (int r = 0; r < 4; ++r) {
    float e = __expf(sv[t + r * 256] - mx);
    sv[t + r * 256] = e;
    lsum += e;
  }
  red[t] = lsum;
  __syncthreads();
  for (int s = 128; s > 0; s >>= 1) { if (t < s) red[t] += red[t + s]; __syncthreads(); }
  float inv = 1.0f / red[0];
  __syncthreads();
  if (t < K_) gath[t] = sv[tki[b * K_ + t]] * inv;
  __syncthreads();
  red[t] = (t < K_) ? gath[t] : 0.0f;
  __syncthreads();
  for (int s = 128; s > 0; s >>= 1) { if (t < s) red[t] += red[t + s]; __syncthreads(); }
  if (t < K_) selo[b * K_ + t] = gath[t] / (red[0] + 1e-12f);
}

// ---------------------------------------------------------------------------
// Host-side orchestration.
// ---------------------------------------------------------------------------
extern "C" void kernel_launch(void* const* d_in, const int* in_sizes, int n_in,
                              void* d_out, int out_size, void* d_ws, size_t ws_size,
                              hipStream_t stream) {
  const float* x       = (const float*)d_in[0];
  const float* gumbel  = (const float*)d_in[1];
  const float* emb_W   = (const float*)d_in[2];
  const float* emb_b   = (const float*)d_in[3];
  const float* Wq      = (const float*)d_in[4];
  const float* Wk      = (const float*)d_in[5];
  const float* Wv      = (const float*)d_in[6];
  const float* Wo      = (const float*)d_in[7];
  const float* ln1_g   = (const float*)d_in[8];
  const float* ln1_b   = (const float*)d_in[9];
  const float* ln2_g   = (const float*)d_in[10];
  const float* ln2_b   = (const float*)d_in[11];
  const float* ffn_W1  = (const float*)d_in[12];
  const float* ffn_b1  = (const float*)d_in[13];
  const float* ffn_W2  = (const float*)d_in[14];
  const float* ffn_b2  = (const float*)d_in[15];
  const float* sha_Wq  = (const float*)d_in[16];
  const float* sha_Wk  = (const float*)d_in[17];
  const float* alloc_W1 = (const float*)d_in[18];
  const float* alloc_b1 = (const float*)d_in[19];
  const float* alloc_W2 = (const float*)d_in[20];
  const float* alloc_b2 = (const float*)d_in[21];
  (void)in_sizes; (void)n_in; (void)out_size; (void)ws_size;

  const size_t BN  = (size_t)B_ * N_;       // 65536 rows
  const size_t HSZ = BN * D_;               // 16,777,216 elems

  // ---- fp32 region ----
  float* h      = (float*)d_ws;             // [BN, D] master activations
  float* tbuf   = h + HSZ;                  // [BN, D] pre-LN sums
  float* kvbuf  = tbuf + HSZ;               // [BN, D]
  float* qv     = kvbuf + HSZ;              // [B, D]
  float* logits = qv + (size_t)B_ * D_;     // [B, N]
  int*   tki    = (int*)(logits + (size_t)B_ * N_);  // [B, K]

  // ---- bf16 region (256B aligned) ----
  bf16_t* bp = (bf16_t*)(((size_t)(tki + (size_t)B_ * K_) + 255) & ~(size_t)255);
  bf16_t* h_bf   = bp;               bp += HSZ;        // activations, bf16 image
  bf16_t* big_bf = bp;               bp += 4 * HSZ;    // q|k|v|o, later ffn
  bf16_t* qbf = big_bf;
  bf16_t* kbf = big_bf + HSZ;
  bf16_t* vbf = big_bf + 2 * HSZ;
  bf16_t* obf = big_bf + 3 * HSZ;
  bf16_t* ffn_bf = big_bf;                            // [BN, DFF] reuses region
  bf16_t* hmean_bf = bp;             bp += (size_t)B_ * D_;
  bf16_t* amid_bf  = bp;             bp += (size_t)B_ * D_;
  // bf16 weights (converted once per launch)
  bf16_t* wq_bf = bp;  bp += (size_t)L_ * D_ * D_;
  bf16_t* wk_bf = bp;  bp += (size_t)L_ * D_ * D_;
  bf16_t* wv_bf = bp;  bp += (size_t)L_ * D_ * D_;
  bf16_t* wo_bf = bp;  bp += (size_t)L_ * D_ * D_;
  bf16_t* w1_bf = bp;  bp += (size_t)L_ * D_ * DFF_;
  bf16_t* w2_bf = bp;  bp += (size_t)L_ * DFF_ * D_;
  bf16_t* shaq_bf = bp; bp += (size_t)D_ * D_;
  bf16_t* shak_bf = bp; bp += (size_t)D_ * D_;
  bf16_t* al1_bf = bp; bp += (size_t)D_ * D_;
  bf16_t* al2_bf = bp; bp += (size_t)D_ * DFF_;

  float* out_w   = (float*)d_out;            // [B, N]
  float* out_idx = out_w + BN;               // [B, K] (as float)
  float* out_sel = out_idx + (size_t)B_ * K_;  // [B, K]

  const dim3 blk(256);
  const dim3 gD(D_ / GBN, (unsigned)((BN + GBM - 1) / GBM));    // (2, 512)
  const dim3 gF(DFF_ / GBN, (unsigned)((BN + GBM - 1) / GBM));  // (8, 512)
  const dim3 gS(D_ / GBN, 1);
  const dim3 gSF(DFF_ / GBN, 1);

  // 0) Weight conversion fp32 -> bf16.
  const int nLDD = L_ * D_ * D_, nLDF = L_ * D_ * DFF_, nDD = D_ * D_, nDF = D_ * DFF_;
  cvt_kernel<<<(nLDD + 255) / 256, blk, 0, stream>>>(Wq, wq_bf, nLDD);
  cvt_kernel<<<(nLDD + 255) / 256, blk, 0, stream>>>(Wk, wk_bf, nLDD);
  cvt_kernel<<<(nLDD + 255) / 256, blk, 0, stream>>>(Wv, wv_bf, nLDD);
  cvt_kernel<<<(nLDD + 255) / 256, blk, 0, stream>>>(Wo, wo_bf, nLDD);
  cvt_kernel<<<(nLDF + 255) / 256, blk, 0, stream>>>(ffn_W1, w1_bf, nLDF);
  cvt_kernel<<<(nLDF + 255) / 256, blk, 0, stream>>>(ffn_W2, w2_bf, nLDF);
  cvt_kernel<<<(nDD + 255) / 256, blk, 0, stream>>>(sha_Wq, shaq_bf, nDD);
  cvt_kernel<<<(nDD + 255) / 256, blk, 0, stream>>>(sha_Wk, shak_bf, nDD);
  cvt_kernel<<<(nDD + 255) / 256, blk, 0, stream>>>(alloc_W1, al1_bf, nDD);
  cvt_kernel<<<(nDF + 255) / 256, blk, 0, stream>>>(alloc_W2, al2_bf, nDF);

  // 1) Embedding (fp32 master + bf16 image).
  embed_kernel<<<(unsigned)BN, blk, 0, stream>>>(x, emb_W, emb_b, h, h_bf);

  // 2) Transformer layers.
  for (int i = 0; i < L_; ++i) {
    const bf16_t* wq = wq_bf + (size_t)i * D_ * D_;
    const bf16_t* wk = wk_bf + (size_t)i * D_ * D_;
    const bf16_t* wv = wv_bf + (size_t)i * D_ * D_;
    const bf16_t* wo = wo_bf + (size_t)i * D_ * D_;
    // qkv: bf16-only outputs; q pre-scaled by 1/sqrt(DH)=0.125.
    gemm_wmma_kernel<<<gD, blk, 0, stream>>>(h_bf, wq, nullptr, nullptr, nullptr, qbf,
                                             (int)BN, D_, D_, 0, 0.125f);
    gemm_wmma_kernel<<<gD, blk, 0, stream>>>(h_bf, wk, nullptr, nullptr, nullptr, kbf,
                                             (int)BN, D_, D_, 0, 1.0f);
    gemm_wmma_kernel<<<gD, blk, 0, stream>>>(h_bf, wv, nullptr, nullptr, nullptr, vbf,
                                             (int)BN, D_, D_, 0, 1.0f);
    attn_kernel<<<B_ * H_ * (N_ / 128), blk, 0, stream>>>(qbf, kbf, vbf, obf);
    gemm_wmma_kernel<<<gD, blk, 0, stream>>>(obf, wo, nullptr, h, tbuf, nullptr,
                                             (int)BN, D_, D_, 0, 1.0f);
    ln_kernel<<<(unsigned)BN, blk, 0, stream>>>(tbuf, ln1_g + i * D_, ln1_b + i * D_, h, h_bf);
    gemm_wmma_kernel<<<gF, blk, 0, stream>>>(h_bf, w1_bf + (size_t)i * D_ * DFF_,
                                             ffn_b1 + (size_t)i * DFF_, nullptr,
                                             nullptr, ffn_bf, (int)BN, D_, DFF_, 1, 1.0f);
    gemm_wmma_kernel<<<gD, blk, 0, stream>>>(ffn_bf, w2_bf + (size_t)i * DFF_ * D_,
                                             ffn_b2 + (size_t)i * D_, h, tbuf, nullptr,
                                             (int)BN, DFF_, D_, 0, 1.0f);
    ln_kernel<<<(unsigned)BN, blk, 0, stream>>>(tbuf, ln2_g + i * D_, ln2_b + i * D_, h, h_bf);
  }

  // 3) Pool + shared-attention scores + gumbel softmax.
  mean_kernel<<<B_, blk, 0, stream>>>(h, hmean_bf);
  gemm_wmma_kernel<<<gS, blk, 0, stream>>>(hmean_bf, shaq_bf, nullptr, nullptr, qv, nullptr,
                                           B_, D_, D_, 0, 1.0f);
  gemm_wmma_kernel<<<gD, blk, 0, stream>>>(h_bf, shak_bf, nullptr, nullptr, kvbuf, nullptr,
                                           (int)BN, D_, D_, 0, 1.0f);
  score_softmax_kernel<<<B_, blk, 0, stream>>>(qv, kvbuf, gumbel, out_w);

  // 4) top-K.
  topk_kernel<<<B_, blk, 0, stream>>>(out_w, out_idx, tki);

  // 5) alloc MLP + softmax + gather/renorm.
  gemm_wmma_kernel<<<gS, blk, 0, stream>>>(hmean_bf, al1_bf, alloc_b1, nullptr, nullptr, amid_bf,
                                           B_, D_, D_, 1, 1.0f);
  gemm_wmma_kernel<<<gSF, blk, 0, stream>>>(amid_bf, al2_bf, alloc_b2, nullptr, logits, nullptr,
                                            B_, D_, DFF_, 0, 1.0f);
  sel_kernel<<<B_, blk, 0, stream>>>(logits, tki, out_sel);
}
